// FractalBlock_64750926954826
// MI455X (gfx1250) — matrix operand order repaired and checked by reading, not compiled, not run on hardware
//
#include <hip/hip_runtime.h>

#define DIM 768
#define HEADS 12
#define HEAD_DIM 64
#define MLP_HID 3072
#define SEQ 1024
#define NBATCH 4
#define TOKENS (NBATCH * SEQ)          // 4096
#define BH (NBATCH * HEADS)            // 48
#define QK_SCALE 0.125f                // 64^-0.5
#define LN_EPS 1e-5f

typedef __attribute__((ext_vector_type(16))) _Float16 v16h;
typedef __attribute__((ext_vector_type(8)))  _Float16 v8h;
typedef __attribute__((ext_vector_type(8)))  float    v8f;
typedef int v4i32 __attribute__((vector_size(16)));

#define GLOBAL_AS __attribute__((address_space(1)))
#define LDS_AS    __attribute__((address_space(3)))

// Use the CDNA5 async global->LDS copy path when the toolchain exposes it.
#if defined(__HIP_DEVICE_COMPILE__) && \
    __has_builtin(__builtin_amdgcn_global_load_async_to_lds_b128) && \
    __has_builtin(__builtin_amdgcn_s_wait_asynccnt)
#define USE_ASYNC_LDS 1
#else
#define USE_ASYNC_LDS 0
#endif

// ---------------------------------------------------------------------------
// Weight convert: W[K,N] fp32 (row major) -> WT[N,K] f16 (row major)
// ---------------------------------------------------------------------------
__global__ void k_wt_f16T(const float* __restrict__ W, _Float16* __restrict__ WT,
                          int K, int N) {
  long idx = (long)blockIdx.x * 256 + threadIdx.x;
  if (idx >= (long)K * N) return;
  int n = (int)(idx % N);
  int k = (int)(idx / N);
  WT[(long)n * K + k] = (_Float16)W[idx];
}

// ---------------------------------------------------------------------------
// LayerNorm over DIM=768, one block (256 thr) per token, f16 output
// ---------------------------------------------------------------------------
__global__ void k_ln(const float* __restrict__ x, const float* __restrict__ g,
                     const float* __restrict__ b, _Float16* __restrict__ out) {
  __shared__ float red[256];
  int tid = threadIdx.x;
  long t = blockIdx.x;
  const float* row = x + t * DIM;
  float v[3];
  float s = 0.f;
#pragma unroll
  for (int i = 0; i < 3; ++i) { v[i] = row[tid + i * 256]; s += v[i]; }
  red[tid] = s; __syncthreads();
  for (int o = 128; o > 0; o >>= 1) { if (tid < o) red[tid] += red[tid + o]; __syncthreads(); }
  float mu = red[0] * (1.0f / DIM);
  __syncthreads();
  float s2 = 0.f;
#pragma unroll
  for (int i = 0; i < 3; ++i) { float d = v[i] - mu; s2 += d * d; }
  red[tid] = s2; __syncthreads();
  for (int o = 128; o > 0; o >>= 1) { if (tid < o) red[tid] += red[tid + o]; __syncthreads(); }
  float rstd = rsqrtf(red[0] * (1.0f / DIM) + LN_EPS);
#pragma unroll
  for (int i = 0; i < 3; ++i) {
    int c = tid + i * 256;
    out[t * DIM + c] = (_Float16)((v[i] - mu) * rstd * g[c] + b[c]);
  }
}

// ---------------------------------------------------------------------------
// A-fragment loaders (16x32 slice, contiguous k = 16*grp + i per lane half). f16 source or fp32
// source converted to f16 in registers.
// ---------------------------------------------------------------------------
__device__ inline v16h load_afrag(const _Float16* __restrict__ arow, int k0, int grp) {
  v8h a0 = *(const v8h*)(arow + k0 + grp * 16);
  v8h a1 = *(const v8h*)(arow + k0 + grp * 16 + 8);
  v16h af;
#pragma unroll
  for (int i = 0; i < 8; ++i) { af[i] = a0[i]; af[i + 8] = a1[i]; }
  return af;
}

__device__ inline v16h load_afrag(const float* __restrict__ arow, int k0, int grp) {
  v8f a0 = *(const v8f*)(arow + k0 + grp * 16);
  v8f a1 = *(const v8f*)(arow + k0 + grp * 16 + 8);
  v16h af;
#pragma unroll
  for (int i = 0; i < 8; ++i) { af[i] = (_Float16)a0[i]; af[i + 8] = (_Float16)a1[i]; }
  return af;
}

// ---------------------------------------------------------------------------
// WMMA GEMM: C[M,N] = A[M,K] @ BT[N,K]^T + bias, fp32 out.
// Block = 256 thr = 8 waves. Block tile 128(M) x 64(N); each wave owns a
// 16x64 strip -> 4 accumulators, one shared A fragment per k-step.
// B tile (64x32 f16 = 4KB) is block-shared: double-buffered in LDS via
// GLOBAL_LOAD_ASYNC_TO_LDS_B128 (ASYNCcnt) when available, else direct
// global b128 loads. grid = (N/64, M/128, Z), batched via element strides.
// ---------------------------------------------------------------------------
template <typename AT>
__global__ __launch_bounds__(256)
void k_gemm(const AT* __restrict__ A, const _Float16* __restrict__ BT,
            const float* __restrict__ bias, float* __restrict__ C,
            int M, int N, int K,
            long long strideA, long long strideB, long long strideC) {
  int lane = threadIdx.x & 31;
  int wave = threadIdx.x >> 5;          // 0..7
  int l16 = lane & 15;
  int grp = lane >> 4;
  int tnBase = blockIdx.x * 64;
  int mRow = blockIdx.y * 128 + wave * 16;

  const AT* Az = A + (long long)blockIdx.z * strideA;
  const _Float16* Bz = BT + (long long)blockIdx.z * strideB;
  float* Cz = C + (long long)blockIdx.z * strideC;

  const AT* arow = Az + (long long)(mRow + l16) * K;
  v8f acc[4] = {};

#if USE_ASYNC_LDS
  __shared__ __align__(32) _Float16 bl[2][64 * 32];   // 2 x 4KB double buffer
  const _Float16* gB = Bz + (long long)tnBase * K;
  int cr = threadIdx.x >> 2;            // row 0..63
  int cs = threadIdx.x & 3;             // 16B segment 0..3
  const _Float16* gsrc = gB + (long long)cr * K + cs * 8;
  _Float16* ldst0 = &bl[0][cr * 32 + cs * 8];
  _Float16* ldst1 = &bl[1][cr * 32 + cs * 8];

  // prologue: stage k0 = 0 into buffer 0
  __builtin_amdgcn_global_load_async_to_lds_b128(
      (GLOBAL_AS v4i32*)gsrc, (LDS_AS v4i32*)ldst0, 0, 0);

  int nk = K >> 5;
  for (int s = 0; s < nk; ++s) {
    int k0 = s << 5;
    int cur = s & 1;
    if (s + 1 < nk) {
      __builtin_amdgcn_global_load_async_to_lds_b128(
          (GLOBAL_AS v4i32*)(gsrc + k0 + 32),
          (LDS_AS v4i32*)(cur ? ldst0 : ldst1), 0, 0);
      __builtin_amdgcn_s_wait_asynccnt(1);   // async loads retire in order
      __builtin_prefetch((const void*)(arow + k0 + 32), 0, 0);
    } else {
      __builtin_amdgcn_s_wait_asynccnt(0);
    }
    __syncthreads();

    v16h af = load_afrag(arow, k0, grp);
    const _Float16* bb = &bl[cur][0];
#pragma unroll
    for (int j = 0; j < 4; ++j) {
      v16h bf = *(const v16h*)(bb + (j * 16 + l16) * 32 + grp * 16);
      acc[j] = __builtin_amdgcn_wmma_f32_16x16x32_f16(false, af, false, bf,
                                                      (short)0, acc[j], false, false);
    }
    __syncthreads();
  }
#else
  for (int k0 = 0; k0 < K; k0 += 32) {
    if (k0 + 32 < K) __builtin_prefetch((const void*)(arow + k0 + 32), 0, 0);
    v16h af = load_afrag(arow, k0, grp);
#pragma unroll
    for (int j = 0; j < 4; ++j) {
      const _Float16* brow = Bz + (long long)(tnBase + j * 16 + l16) * K;
      v16h bf = *(const v16h*)(brow + k0 + grp * 16);
      acc[j] = __builtin_amdgcn_wmma_f32_16x16x32_f16(false, af, false, bf,
                                                      (short)0, acc[j], false, false);
    }
  }
#endif

  int nCol = tnBase + l16;
#pragma unroll
  for (int j = 0; j < 4; ++j) {
    float bv = bias ? bias[nCol + j * 16] : 0.0f;
#pragma unroll
    for (int r = 0; r < 8; ++r) {
      int m = mRow + r + grp * 8;
      Cz[(long long)m * N + nCol + j * 16] = acc[j][r] + bv;
    }
  }
}

// ---------------------------------------------------------------------------
// Split QKV fp32 [4096, 2304] -> q16 (scaled), k16 [bh,n,64], vT16 [bh,64,n]
// ---------------------------------------------------------------------------
__global__ void k_split_qkv(const float* __restrict__ qkv, _Float16* __restrict__ q,
                            _Float16* __restrict__ k, _Float16* __restrict__ vT) {
  long idx = (long)blockIdx.x * 256 + threadIdx.x;
  if (idx >= (long)TOKENS * 3 * DIM) return;
  int c = (int)(idx % (3 * DIM));
  long t = idx / (3 * DIM);
  int b = (int)(t >> 10);
  int n = (int)(t & 1023);
  int which = c / DIM;
  int cc = c % DIM;
  int h = cc >> 6;
  int d = cc & 63;
  float val = qkv[idx];
  long bh = (long)b * HEADS + h;
  if (which == 0)      q[(bh * SEQ + n) * HEAD_DIM + d] = (_Float16)(val * QK_SCALE);
  else if (which == 1) k[(bh * SEQ + n) * HEAD_DIM + d] = (_Float16)val;
  else                 vT[(bh * HEAD_DIM + d) * SEQ + n] = (_Float16)val;
}

// ---------------------------------------------------------------------------
// Softmax over rows of 1024, in place. One block (256 thr) per row.
// ---------------------------------------------------------------------------
__global__ void k_softmax(float* __restrict__ S) {
  __shared__ float red[256];
  int tid = threadIdx.x;
  long row = blockIdx.x;
  float* p = S + row * SEQ;
  float v[4];
  float mx = -3.402823466e38f;
#pragma unroll
  for (int i = 0; i < 4; ++i) { v[i] = p[tid + i * 256]; mx = fmaxf(mx, v[i]); }
  red[tid] = mx; __syncthreads();
  for (int o = 128; o > 0; o >>= 1) { if (tid < o) red[tid] = fmaxf(red[tid], red[tid + o]); __syncthreads(); }
  mx = red[0];
  __syncthreads();
  float s = 0.f;
#pragma unroll
  for (int i = 0; i < 4; ++i) { v[i] = expf(v[i] - mx); s += v[i]; }
  red[tid] = s; __syncthreads();
  for (int o = 128; o > 0; o >>= 1) { if (tid < o) red[tid] += red[tid + o]; __syncthreads(); }
  float inv = 1.0f / red[0];
#pragma unroll
  for (int i = 0; i < 4; ++i) p[tid + i * 256] = v[i] * inv;
}

// ---------------------------------------------------------------------------
// Merge heads: av fp32 [bh, n, 64] -> f16 [token, 768]
// ---------------------------------------------------------------------------
__global__ void k_merge_heads(const float* __restrict__ av, _Float16* __restrict__ out) {
  long idx = (long)blockIdx.x * 256 + threadIdx.x;
  if (idx >= (long)BH * SEQ * HEAD_DIM) return;
  int d = (int)(idx & 63);
  long r = idx >> 6;
  int n = (int)(r & 1023);
  r >>= 10;
  int h = (int)(r % HEADS);
  int b = (int)(r / HEADS);
  out[((long)(b * SEQ + n)) * DIM + h * HEAD_DIM + d] = (_Float16)av[idx];
}

// ---------------------------------------------------------------------------
// Elementwise: exact GELU -> f16 ; residual add ; 0.5 blend
// ---------------------------------------------------------------------------
__global__ void k_gelu(const float* __restrict__ in, _Float16* __restrict__ out, long n) {
  long idx = (long)blockIdx.x * 256 + threadIdx.x;
  if (idx >= n) return;
  float x = in[idx];
  out[idx] = (_Float16)(0.5f * x * (1.0f + erff(x * 0.70710678118654752f)));
}

__global__ void k_add(float* __restrict__ x, const float* __restrict__ d, long n) {
  long idx = (long)blockIdx.x * 256 + threadIdx.x;
  if (idx < n) x[idx] += d[idx];
}

__global__ void k_blend(float* __restrict__ x, const float* __restrict__ xs, long n) {
  long idx = (long)blockIdx.x * 256 + threadIdx.x;
  if (idx < n) x[idx] = 0.5f * (x[idx] + xs[idx]);
}

// ---------------------------------------------------------------------------
// Host-side orchestration
// ---------------------------------------------------------------------------
namespace {

struct Ws {
  float* x;        // [4096,768]
  float* xsave;    // [4096,768]
  float* g32;      // [4096,3072] scratch GEMM output (also holds qkv 4096x2304)
  float* av32;     // [48,1024,64]
  _Float16* h16;   // [4096,3072] f16 activations (LN out / gelu out / attn out)
  _Float16* q16;   // [48,1024,64]
  _Float16* k16;   // [48,1024,64]
  _Float16* vT16;  // [48,64,1024]
  _Float16* qkvT[2];
  _Float16* projT[2];
  _Float16* w1T[2];
  _Float16* w2T[2];
};

inline void* bump(char*& p, size_t bytes) {
  void* r = (void*)p;
  p += (bytes + 255) & ~(size_t)255;
  return r;
}

inline dim3 gemm_grid(int M, int N, int Z) { return dim3(N / 64, M / 128, Z); }

void launch_gemm16(const _Float16* A, const _Float16* BT, const float* bias, float* C,
                   int M, int N, int K, long long sA, long long sB, long long sC,
                   int Z, hipStream_t s) {
  k_gemm<_Float16><<<gemm_grid(M, N, Z), 256, 0, s>>>(A, BT, bias, C, M, N, K, sA, sB, sC);
}

void run_attn(const Ws& w, const float* ln_g, const float* ln_b,
              const _Float16* qkvT, const float* qkv_b,
              const _Float16* projT, const float* proj_b,
              float* S, hipStream_t s) {
  // h16 = LN1(x)
  k_ln<<<TOKENS, 256, 0, s>>>(w.x, ln_g, ln_b, w.h16);
  // g32 = h16 @ qkv_w + qkv_b   [4096, 2304]
  launch_gemm16(w.h16, qkvT, qkv_b, w.g32, TOKENS, 3 * DIM, DIM, 0, 0, 0, 1, s);
  // split into q (scaled), k, v^T
  k_split_qkv<<<(int)(((long)TOKENS * 3 * DIM + 255) / 256), 256, 0, s>>>(w.g32, w.q16, w.k16, w.vT16);
  // S = (q*scale) @ k^T  -> straight into the attention-map output region
  launch_gemm16(w.q16, w.k16, nullptr, S, SEQ, SEQ, HEAD_DIM,
                (long long)SEQ * HEAD_DIM, (long long)SEQ * HEAD_DIM,
                (long long)SEQ * SEQ, BH, s);
  // softmax in place
  k_softmax<<<BH * SEQ, 256, 0, s>>>(S);
  // av = P @ v  (P fp32 read from d_out, converted to f16 in regs)
  k_gemm<float><<<gemm_grid(SEQ, HEAD_DIM, BH), 256, 0, s>>>(
      S, w.vT16, nullptr, w.av32, SEQ, HEAD_DIM, SEQ,
      (long long)SEQ * SEQ, (long long)HEAD_DIM * SEQ, (long long)SEQ * HEAD_DIM);
  // merge heads -> h16 [4096,768] f16
  k_merge_heads<<<(int)(((long)BH * SEQ * HEAD_DIM + 255) / 256), 256, 0, s>>>(w.av32, w.h16);
  // g32 = attn_out @ proj_w + proj_b
  launch_gemm16(w.h16, projT, proj_b, w.g32, TOKENS, DIM, DIM, 0, 0, 0, 1, s);
  // x += g32
  k_add<<<(TOKENS * DIM) / 256, 256, 0, s>>>(w.x, w.g32, (long)TOKENS * DIM);
}

void run_mlp(const Ws& w, const float* ln_g, const float* ln_b,
             const _Float16* w1T, const float* b1,
             const _Float16* w2T, const float* b2, hipStream_t s) {
  k_ln<<<TOKENS, 256, 0, s>>>(w.x, ln_g, ln_b, w.h16);
  launch_gemm16(w.h16, w1T, b1, w.g32, TOKENS, MLP_HID, DIM, 0, 0, 0, 1, s);
  k_gelu<<<(TOKENS * MLP_HID) / 256, 256, 0, s>>>(w.g32, w.h16, (long)TOKENS * MLP_HID);
  launch_gemm16(w.h16, w2T, b2, w.g32, TOKENS, DIM, MLP_HID, 0, 0, 0, 1, s);
  k_add<<<(TOKENS * DIM) / 256, 256, 0, s>>>(w.x, w.g32, (long)TOKENS * DIM);
}

} // namespace

extern "C" void kernel_launch(void* const* d_in, const int* in_sizes, int n_in,
                              void* d_out, int out_size, void* d_ws, size_t ws_size,
                              hipStream_t stream) {
  (void)in_sizes; (void)n_in; (void)out_size; (void)ws_size;

  const float* in_x = (const float*)d_in[0];
  // per-depth inputs: qkv_w, qkv_b, proj_w, proj_b, ln1_g, ln1_b, ln2_g, ln2_b,
  //                   mlp_w1, mlp_b1, mlp_w2, mlp_b2
  const float* P[2][12];
  for (int d = 0; d < 2; ++d)
    for (int i = 0; i < 12; ++i)
      P[d][i] = (const float*)d_in[1 + d * 12 + i];

  // ---- workspace layout ----
  char* p = (char*)d_ws;
  Ws w;
  w.x     = (float*)bump(p, (size_t)TOKENS * DIM * 4);
  w.xsave = (float*)bump(p, (size_t)TOKENS * DIM * 4);
  w.g32   = (float*)bump(p, (size_t)TOKENS * MLP_HID * 4);
  w.av32  = (float*)bump(p, (size_t)BH * SEQ * HEAD_DIM * 4);
  w.h16   = (_Float16*)bump(p, (size_t)TOKENS * MLP_HID * 2);
  w.q16   = (_Float16*)bump(p, (size_t)BH * SEQ * HEAD_DIM * 2);
  w.k16   = (_Float16*)bump(p, (size_t)BH * SEQ * HEAD_DIM * 2);
  w.vT16  = (_Float16*)bump(p, (size_t)BH * SEQ * HEAD_DIM * 2);
  for (int d = 0; d < 2; ++d) {
    w.qkvT[d]  = (_Float16*)bump(p, (size_t)DIM * 3 * DIM * 2);
    w.projT[d] = (_Float16*)bump(p, (size_t)DIM * DIM * 2);
    w.w1T[d]   = (_Float16*)bump(p, (size_t)DIM * MLP_HID * 2);
    w.w2T[d]   = (_Float16*)bump(p, (size_t)MLP_HID * DIM * 2);
  }

  float* out = (float*)d_out;
  float* attn0 = out + (size_t)TOKENS * DIM;
  float* attn1 = attn0 + (size_t)BH * SEQ * SEQ;

  // ---- residual stream init ----
  (void)hipMemcpyAsync(w.x, in_x, (size_t)TOKENS * DIM * 4, hipMemcpyDeviceToDevice, stream);

  // ---- convert weights to f16, transposed [N,K] ----
  for (int d = 0; d < 2; ++d) {
    k_wt_f16T<<<(int)(((long)DIM * 3 * DIM + 255) / 256), 256, 0, stream>>>(P[d][0], w.qkvT[d], DIM, 3 * DIM);
    k_wt_f16T<<<(int)(((long)DIM * DIM + 255) / 256), 256, 0, stream>>>(P[d][2], w.projT[d], DIM, DIM);
    k_wt_f16T<<<(int)(((long)DIM * MLP_HID + 255) / 256), 256, 0, stream>>>(P[d][8], w.w1T[d], DIM, MLP_HID);
    k_wt_f16T<<<(int)(((long)MLP_HID * DIM + 255) / 256), 256, 0, stream>>>(P[d][10], w.w2T[d], MLP_HID, DIM);
  }

  // ---- level 0 attention ----
  run_attn(w, P[0][4], P[0][5], w.qkvT[0], P[0][1], w.projT[0], P[0][3], attn0, stream);
  // save x for blend
  (void)hipMemcpyAsync(w.xsave, w.x, (size_t)TOKENS * DIM * 4, hipMemcpyDeviceToDevice, stream);
  // ---- level 1 (full block) ----
  run_attn(w, P[1][4], P[1][5], w.qkvT[1], P[1][1], w.projT[1], P[1][3], attn1, stream);
  run_mlp(w, P[1][6], P[1][7], w.w1T[1], P[1][9], w.w2T[1], P[1][11], stream);
  // blend x = 0.5*xsave + 0.5*x_sub
  k_blend<<<(TOKENS * DIM) / 256, 256, 0, stream>>>(w.x, w.xsave, (long)TOKENS * DIM);
  // ---- level 0 MLP ----
  run_mlp(w, P[0][6], P[0][7], w.w1T[0], P[0][9], w.w2T[0], P[0][11], stream);

  // ---- final output ----
  (void)hipMemcpyAsync(out, w.x, (size_t)TOKENS * DIM * 4, hipMemcpyDeviceToDevice, stream);
}
